// point_transformer_network_76158360093250
// MI455X (gfx1250) — compile-verified, matrix operand order checked by reading, not run on hardware
//
#include <hip/hip_runtime.h>

#define BATCH 2
#define KNB 16

typedef __attribute__((ext_vector_type(16))) _Float16 v16h;
typedef __attribute__((ext_vector_type(8)))  float    v8f;

static inline int cdiv(int a, int b) { return (a + b - 1) / b; }

// ------------------------------------------------------------------
// Weight pre-pack: W (K x N, f32 row-major) -> f16 WMMA B-fragments,
// layout [ntile][kc][lane][16] so each lane reads its v16h contiguously.
// Element j of lane's fragment covers kk = kc*32 + j + (j<8?0:8) + (lane>>4)*8,
// col = ntile*16 + (lane&15); out-of-range K/N zero-padded here (no guards in
// the hot kernel).
// ------------------------------------------------------------------
__global__ void prep_wfrag_kernel(const float* __restrict__ W, _Float16* __restrict__ Wf,
                                  int K, int N, int total)
{
  int t = blockIdx.x * blockDim.x + threadIdx.x;
  if (t >= total) return;
  const int KC = (K + 31) / 32;
  int j = t & 15;
  int lane = (t >> 4) & 31;
  int rest = t >> 9;
  int kc = rest % KC;
  int nt = rest / KC;
  int hi = lane >> 4, r = lane & 15;
  int kk = kc * 32 + j + ((j < 8) ? 0 : 8) + hi * 8;
  int col = nt * 16 + r;
  float w = (kk < K && col < N) ? W[kk * N + col] : 0.0f;
  Wf[t] = (_Float16)w;
}

// ------------------------------------------------------------------
// Generic dense layer Y = relu?(X*W + b) via CDNA5 WMMA f16->f32.
// K is a template parameter: loop fully unrolled, guards compile-time.
// Requires M % 16 == 0 (true for every layer in this network).
// One wave per 16x16 output tile.
// ------------------------------------------------------------------
template <int K>
__global__ void dense_wmma_kernel(const float* __restrict__ X, const _Float16* __restrict__ Wf,
                                  const float* __restrict__ bias, float* __restrict__ Y,
                                  int M, int N, int relu)
{
  constexpr int KC = (K + 31) / 32;
  const int tm = blockIdx.x << 4;
  const int lane = threadIdx.x & 31;
  const int r  = lane & 15;
  const int hi = lane >> 4;
  const int arow = tm + r;
  const float* __restrict__ xrow = X + (size_t)arow * K;
  const _Float16* __restrict__ wb0 = Wf + ((size_t)(blockIdx.y * KC) * 32 + lane) * 16;

  v8f acc = {};
#pragma unroll
  for (int kc = 0; kc < KC; ++kc) {
    const int k0 = kc * 32;
    // B fragment: one contiguous 32-byte load from pre-packed weights
    v16h bv = *(const v16h*)(wb0 + (size_t)kc * 32 * 16);
    v16h av;
    if constexpr ((K & 7) == 0) {
      // Each lane needs two aligned runs of 8 consecutive K-elements.
      const int b1 = k0 + hi * 8;        // j = 0..7  -> kk = b1 + j
      const int b2 = k0 + 16 + hi * 8;   // j = 8..15 -> kk = b2 + (j-8)
      const bool v1 = (b1 < K);
      const bool v2 = (b2 < K);
      const int o1 = v1 ? b1 : 0;
      const int o2 = v2 ? b2 : 0;
      const float m1 = v1 ? 1.0f : 0.0f;
      const float m2 = v2 ? 1.0f : 0.0f;
      const float4 A0 = *(const float4*)(xrow + o1);
      const float4 A1 = *(const float4*)(xrow + o1 + 4);
      const float4 B0 = *(const float4*)(xrow + o2);
      const float4 B1 = *(const float4*)(xrow + o2 + 4);
      float g[16] = {A0.x, A0.y, A0.z, A0.w, A1.x, A1.y, A1.z, A1.w,
                     B0.x, B0.y, B0.z, B0.w, B1.x, B1.y, B1.z, B1.w};
#pragma unroll
      for (int j = 0; j < 16; ++j) {
        const float m = (j < 8) ? m1 : m2;
        av[j] = (_Float16)(g[j] * m);
      }
    } else {
      // Small-K path (K = 3,4,6): clamped scalar loads + selects, branch-free.
#pragma unroll
      for (int j = 0; j < 16; ++j) {
        const int kk = k0 + j + ((j < 8) ? 0 : 8) + hi * 8;
        const int kcl = (kk < K) ? kk : 0;
        float a = xrow[kcl];
        av[j] = (_Float16)((kk < K) ? a : 0.0f);
      }
    }
    acc = __builtin_amdgcn_wmma_f32_16x16x32_f16(false, av, false, bv,
                                                 (short)0, acc, false, false);
  }

  const int bcol = (blockIdx.y << 4) + r;
  if (bcol < N) {
    const float bb = bias[bcol];
#pragma unroll
    for (int i = 0; i < 8; ++i) {
      const int row = tm + (hi << 3) + i;
      float v = acc[i] + bb;
      if (relu) v = fmaxf(v, 0.0f);
      Y[(size_t)row * N + bcol] = v;
    }
  }
}

// ---------------- split points -> pos(3), feat(6) ----------------
__global__ void split_points_kernel(const float* __restrict__ pts, float* __restrict__ pos,
                                    float* __restrict__ feat, int total /*B*N*/)
{
  int t = blockIdx.x * blockDim.x + threadIdx.x;
  if (t >= total) return;
  const float* p = pts + t * 9;
  pos[t * 3 + 0] = p[0]; pos[t * 3 + 1] = p[1]; pos[t * 3 + 2] = p[2];
#pragma unroll
  for (int c = 0; c < 6; ++c) feat[t * 6 + c] = p[3 + c];
}

// ---------------- Farthest point sampling ----------------
// One block per batch; mind[] in LDS; serial scan with block argmax
// (first-index tie-break, matching jnp.argmax).
__global__ void fps_kernel(const float* __restrict__ pos, int N, int m, int* __restrict__ ind)
{
  extern __shared__ float smem[];
  float* mind = smem;                 // N floats
  float* redv = mind + N;             // blockDim floats
  int*   redi = (int*)(redv + blockDim.x);
  __shared__ int s_last;
  __shared__ float s_lp[3];
  const int b = blockIdx.x;
  const float* posb = pos + (size_t)b * N * 3;
  for (int i = threadIdx.x; i < N; i += blockDim.x) mind[i] = 1e10f;
  if (threadIdx.x == 0) { ind[b * m] = 0; s_last = 0; }
  __syncthreads();
  for (int step = 1; step < m; ++step) {
    if (threadIdx.x == 0) {
      int l = s_last;
      s_lp[0] = posb[l * 3]; s_lp[1] = posb[l * 3 + 1]; s_lp[2] = posb[l * 3 + 2];
    }
    __syncthreads();
    const float lx = s_lp[0], ly = s_lp[1], lz = s_lp[2];
    float bv = -1.0f; int bi = N;
    for (int i = threadIdx.x; i < N; i += blockDim.x) {
      float dx = posb[i * 3] - lx, dy = posb[i * 3 + 1] - ly, dz = posb[i * 3 + 2] - lz;
      float d = fminf(mind[i], dx * dx + dy * dy + dz * dz);
      mind[i] = d;
      if (d > bv) { bv = d; bi = i; }
    }
    redv[threadIdx.x] = bv; redi[threadIdx.x] = bi;
    __syncthreads();
    for (int off = blockDim.x >> 1; off > 0; off >>= 1) {
      if ((int)threadIdx.x < off) {
        float ov = redv[threadIdx.x + off]; int oi = redi[threadIdx.x + off];
        if (ov > redv[threadIdx.x] || (ov == redv[threadIdx.x] && oi < redi[threadIdx.x])) {
          redv[threadIdx.x] = ov; redi[threadIdx.x] = oi;
        }
      }
      __syncthreads();
    }
    if (threadIdx.x == 0) { s_last = redi[0]; ind[b * m + step] = redi[0]; }
    __syncthreads();
  }
}

// ---------------- brute-force kNN (k=16), register top-k per query ----------------
__global__ void knn_kernel(const float* __restrict__ src, const float* __restrict__ query,
                           int S, int Q, int* __restrict__ oidx, float* __restrict__ odist)
{
  const int t = blockIdx.x * blockDim.x + threadIdx.x;
  const int b = blockIdx.y;
  if (t >= Q) return;
  const float* sb = src + (size_t)b * S * 3;
  const float* qb = query + (size_t)b * Q * 3;
  const float qx = qb[t * 3], qy = qb[t * 3 + 1], qz = qb[t * 3 + 2];
  float bd[KNB]; int bi[KNB];
#pragma unroll
  for (int j = 0; j < KNB; ++j) { bd[j] = 3.4e38f; bi[j] = 0; }
  for (int i = 0; i < S; ++i) {
    float dx = sb[i * 3] - qx, dy = sb[i * 3 + 1] - qy, dz = sb[i * 3 + 2] - qz;
    float d = dx * dx + dy * dy + dz * dz;
    if (d < bd[KNB - 1]) {
      bd[KNB - 1] = d; bi[KNB - 1] = i;
#pragma unroll
      for (int j = KNB - 1; j > 0; --j) {
        if (bd[j] < bd[j - 1]) {
          float td = bd[j]; bd[j] = bd[j - 1]; bd[j - 1] = td;
          int ti = bi[j]; bi[j] = bi[j - 1]; bi[j - 1] = ti;
        }
      }
    }
  }
  int* oi = oidx + ((size_t)b * Q + t) * KNB;
  float* od = odist + ((size_t)b * Q + t) * KNB;
#pragma unroll
  for (int j = 0; j < KNB; ++j) {
    oi[j] = bi[j];
    od[j] = sqrtf(fmaxf(fmaxf(bd[j], 0.0f), 1e-12f));
  }
}

// out[b,m,c] = src[b, idx[b,m], c]  (C=3, for sampled positions)
__global__ void gather_points_kernel(const float* __restrict__ src, const int* __restrict__ idx,
                                     float* __restrict__ out, int S, int Mq, int total)
{
  int t = blockIdx.x * blockDim.x + threadIdx.x;
  if (t >= total) return;
  int c = t % 3; int m = (t / 3) % Mq; int b = t / (3 * Mq);
  out[t] = src[((size_t)b * S + idx[b * Mq + m]) * 3 + c];
}

// out[b,q,j,c] = feat[b, nidx[b,q,j], c]
__global__ void gather_nb_kernel(const float* __restrict__ feat, const int* __restrict__ nidx,
                                 float* __restrict__ out, int S, int Q, int C, int total)
{
  int t = blockIdx.x * blockDim.x + threadIdx.x;
  if (t >= total) return;
  int c = t % C; int r = t / C; int j = r % KNB; int q = (r / KNB) % Q; int b = r / (KNB * Q);
  int s = nidx[((size_t)b * Q + q) * KNB + j];
  out[t] = feat[((size_t)b * S + s) * C + c];
}

// pr[b,q,j,c] = pos[b,nidx,c] - pos[b,q,c]
__global__ void gather_pr_kernel(const float* __restrict__ pos, const int* __restrict__ nidx,
                                 float* __restrict__ out, int N, int total)
{
  int t = blockIdx.x * blockDim.x + threadIdx.x;
  if (t >= total) return;
  int c = t % 3; int r = t / 3; int j = r % KNB; int q = (r / KNB) % N; int b = r / (KNB * N);
  int s = nidx[((size_t)b * N + q) * KNB + j];
  out[t] = pos[((size_t)b * N + s) * 3 + c] - pos[((size_t)b * N + q) * 3 + c];
}

// s[b,q,j,c] = q[b,nidx,c] - k[b,q,c] + pe[b,q,j,c]
__global__ void qkpe_kernel(const float* __restrict__ qf, const float* __restrict__ kf,
                            const float* __restrict__ pe, const int* __restrict__ nidx,
                            float* __restrict__ out, int N, int C, int total)
{
  int t = blockIdx.x * blockDim.x + threadIdx.x;
  if (t >= total) return;
  int c = t % C; int r = t / C; int j = r % KNB; int q = (r / KNB) % N; int b = r / (KNB * N);
  int s = nidx[((size_t)b * N + q) * KNB + j];
  out[t] = qf[((size_t)b * N + s) * C + c] - kf[((size_t)b * N + q) * C + c] + pe[t];
}

// out[b,m,c] = max_j x[b,m,j,c]
__global__ void rowmax16_kernel(const float* __restrict__ x, float* __restrict__ out,
                                int C, int total)
{
  int t = blockIdx.x * blockDim.x + threadIdx.x;
  if (t >= total) return;
  int c = t % C; int m = t / C;
  const float* xb = x + (size_t)m * KNB * C + c;
  float v = xb[0];
#pragma unroll
  for (int j = 1; j < KNB; ++j) v = fmaxf(v, xb[j * C]);
  out[t] = v;
}

// out[b,q,c] = sum_j vn[b,q,j,c] * softmax_j(a[b,q,j,c])
__global__ void attn_softmax_sum_kernel(const float* __restrict__ a, const float* __restrict__ vn,
                                        float* __restrict__ out, int Q, int C, int total)
{
  int t = blockIdx.x * blockDim.x + threadIdx.x;
  if (t >= total) return;
  int c = t % C; int q = (t / C) % Q; int b = t / (C * Q);
  const size_t base = ((size_t)b * Q + q) * KNB * C + c;
  const float* ab = a + base;
  const float* vb = vn + base;
  float av[KNB];
  float mx = -3.4e38f;
#pragma unroll
  for (int j = 0; j < KNB; ++j) { av[j] = ab[j * C]; mx = fmaxf(mx, av[j]); }
  float s = 0.0f;
#pragma unroll
  for (int j = 0; j < KNB; ++j) { av[j] = expf(av[j] - mx); s += av[j]; }
  float acc = 0.0f;
#pragma unroll
  for (int j = 0; j < KNB; ++j) acc += vb[j * C] * av[j];
  out[t] = acc / s;
}

// inverse-distance interpolation: out[b,q,c] = sum_j w_j feat[b,nidx_j,c]
__global__ void interp_kernel(const float* __restrict__ feat, const int* __restrict__ nidx,
                              const float* __restrict__ nd, float* __restrict__ out,
                              int S, int Q, int C, int total)
{
  int t = blockIdx.x * blockDim.x + threadIdx.x;
  if (t >= total) return;
  int c = t % C; int q = (t / C) % Q; int b = t / (C * Q);
  const int* ib = nidx + ((size_t)b * Q + q) * KNB;
  const float* db = nd + ((size_t)b * Q + q) * KNB;
  float wsum = 0.0f, acc = 0.0f;
#pragma unroll
  for (int j = 0; j < KNB; ++j) {
    float w = 1.0f / (db[j] + 1e-6f);
    wsum += w;
    acc += w * feat[((size_t)b * S + ib[j]) * C + c];
  }
  out[t] = acc / wsum;
}

__global__ void concat_kernel(const float* __restrict__ A, const float* __restrict__ Bx,
                              float* __restrict__ out, int C1, int C2, int total)
{
  int t = blockIdx.x * blockDim.x + threadIdx.x;
  if (t >= total) return;
  int C = C1 + C2; int c = t % C; int r = t / C;
  out[t] = (c < C1) ? A[(size_t)r * C1 + c] : Bx[(size_t)r * C2 + (c - C1)];
}

__global__ void finalnorm_kernel(const float* __restrict__ x, float* __restrict__ out, int rows)
{
  int t = blockIdx.x * blockDim.x + threadIdx.x;
  if (t >= rows) return;
  const float* xr = x + (size_t)t * 13;
  float s = 0.0f;
#pragma unroll
  for (int i = 0; i < 13; ++i) s += xr[i] * xr[i];
  float inv = 10.0f / sqrtf(s);
#pragma unroll
  for (int i = 0; i < 13; ++i) out[(size_t)t * 13 + i] = xr[i] * inv;
}

// ------------------------------------------------------------------
// Host-side orchestration
// ------------------------------------------------------------------
struct DenseP { const float* W; const float* b; _Float16* Wf; int din, dout; };
struct PTP { DenseP lin1, q, k, v, pos1, pos2, attn1, attn2, lin2; };

struct Bump {
  char* base; size_t off; size_t cap;
  float* f(size_t n) { size_t a = (off + 255) & ~(size_t)255; off = a + n * sizeof(float); return (float*)(base + a); }
  int*   i(size_t n) { size_t a = (off + 255) & ~(size_t)255; off = a + n * sizeof(int);   return (int*)(base + a); }
  _Float16* h(size_t n) { size_t a = (off + 255) & ~(size_t)255; off = a + n * sizeof(_Float16); return (_Float16*)(base + a); }
};

static void launch_dense(const DenseP& p, const float* X, float* Y, int M, int relu,
                         hipStream_t stream)
{
  dim3 g(M / 16, cdiv(p.dout, 16));  // all layers in this net have M % 16 == 0
  switch (p.din) {
    case 3:  dense_wmma_kernel<3 ><<<g, 32, 0, stream>>>(X, p.Wf, p.b, Y, M, p.dout, relu); break;
    case 4:  dense_wmma_kernel<4 ><<<g, 32, 0, stream>>>(X, p.Wf, p.b, Y, M, p.dout, relu); break;
    case 6:  dense_wmma_kernel<6 ><<<g, 32, 0, stream>>>(X, p.Wf, p.b, Y, M, p.dout, relu); break;
    case 8:  dense_wmma_kernel<8 ><<<g, 32, 0, stream>>>(X, p.Wf, p.b, Y, M, p.dout, relu); break;
    case 16: dense_wmma_kernel<16><<<g, 32, 0, stream>>>(X, p.Wf, p.b, Y, M, p.dout, relu); break;
    case 32: dense_wmma_kernel<32><<<g, 32, 0, stream>>>(X, p.Wf, p.b, Y, M, p.dout, relu); break;
    case 40: dense_wmma_kernel<40><<<g, 32, 0, stream>>>(X, p.Wf, p.b, Y, M, p.dout, relu); break;
    case 64: dense_wmma_kernel<64><<<g, 32, 0, stream>>>(X, p.Wf, p.b, Y, M, p.dout, relu); break;
    default: break; // unreachable for this network
  }
}

static void run_down(const DenseP& D, const float* feat, const float* pos, int N, int M,
                     float* outf, float* outp, Bump ws, hipStream_t stream)
{
  int* ind = ws.i((size_t)BATCH * M);
  size_t shmem = sizeof(float) * (size_t)N + 256 * (sizeof(float) + sizeof(int));
  fps_kernel<<<BATCH, 256, shmem, stream>>>(pos, N, M, ind);
  { int total = BATCH * M * 3;
    gather_points_kernel<<<cdiv(total, 256), 256, 0, stream>>>(pos, ind, outp, N, M, total); }
  int* nidx = ws.i((size_t)BATCH * M * KNB);
  float* nd = ws.f((size_t)BATCH * M * KNB);
  knn_kernel<<<dim3(cdiv(M, 128), BATCH), 128, 0, stream>>>(pos, outp, N, M, nidx, nd);
  float* gath = ws.f((size_t)BATCH * M * KNB * D.din);
  { int total = BATCH * M * KNB * D.din;
    gather_nb_kernel<<<cdiv(total, 256), 256, 0, stream>>>(feat, nidx, gath, N, M, D.din, total); }
  float* dl = ws.f((size_t)BATCH * M * KNB * D.dout);
  launch_dense(D, gath, dl, BATCH * M * KNB, 1, stream);
  { int total = BATCH * M * D.dout;
    rowmax16_kernel<<<cdiv(total, 256), 256, 0, stream>>>(dl, outf, D.dout, total); }
}

static void run_pt(const PTP& P, const float* feat, const float* pos, int N,
                   float* out, Bump ws, hipStream_t stream)
{
  const int dim = P.lin1.dout;
  const int rows = BATCH * N;
  const int nb = rows * KNB;
  float* f  = ws.f((size_t)rows * dim); launch_dense(P.lin1, feat, f, rows, 1, stream);
  float* qf = ws.f((size_t)rows * dim); launch_dense(P.q, f, qf, rows, 1, stream);
  float* kf = ws.f((size_t)rows * dim); launch_dense(P.k, f, kf, rows, 1, stream);
  float* vf = ws.f((size_t)rows * dim); launch_dense(P.v, f, vf, rows, 1, stream);
  int* nidx = ws.i((size_t)nb);
  float* nd = ws.f((size_t)nb);
  knn_kernel<<<dim3(cdiv(N, 128), BATCH), 128, 0, stream>>>(pos, pos, N, N, nidx, nd);
  float* pr = ws.f((size_t)nb * 3);
  { int total = nb * 3;
    gather_pr_kernel<<<cdiv(total, 256), 256, 0, stream>>>(pos, nidx, pr, N, total); }
  float* h8 = ws.f((size_t)nb * 8); launch_dense(P.pos1, pr, h8, nb, 1, stream);
  float* pe = ws.f((size_t)nb * dim); launch_dense(P.pos2, h8, pe, nb, 1, stream);
  float* s = ws.f((size_t)nb * dim);
  { int total = nb * dim;
    qkpe_kernel<<<cdiv(total, 256), 256, 0, stream>>>(qf, kf, pe, nidx, s, N, dim, total); }
  float* a4 = ws.f((size_t)nb * 4); launch_dense(P.attn1, s, a4, nb, 1, stream);
  float* a  = ws.f((size_t)nb * dim); launch_dense(P.attn2, a4, a, nb, 1, stream);
  float* vn = ws.f((size_t)nb * dim);
  { int total = nb * dim;
    gather_nb_kernel<<<cdiv(total, 256), 256, 0, stream>>>(vf, nidx, vn, N, N, dim, total); }
  float* po = ws.f((size_t)rows * dim);
  { int total = rows * dim;
    attn_softmax_sum_kernel<<<cdiv(total, 256), 256, 0, stream>>>(a, vn, po, N, dim, total); }
  launch_dense(P.lin2, po, out, rows, 0, stream);
}

static void run_up(const DenseP& D, const float* featC, const float* posC, const float* posF,
                   int S, int Q, float* out, Bump ws, hipStream_t stream)
{
  int* nidx = ws.i((size_t)BATCH * Q * KNB);
  float* nd = ws.f((size_t)BATCH * Q * KNB);
  knn_kernel<<<dim3(cdiv(Q, 128), BATCH), 128, 0, stream>>>(posC, posF, S, Q, nidx, nd);
  float* itp = ws.f((size_t)BATCH * Q * D.din);
  { int total = BATCH * Q * D.din;
    interp_kernel<<<cdiv(total, 256), 256, 0, stream>>>(featC, nidx, nd, itp, S, Q, D.din, total); }
  launch_dense(D, itp, out, BATCH * Q, 1, stream);
}

extern "C" void kernel_launch(void* const* d_in, const int* in_sizes, int n_in,
                              void* d_out, int out_size, void* d_ws, size_t ws_size,
                              hipStream_t stream)
{
  (void)in_sizes; (void)n_in; (void)out_size;
  // Input order: setup_inputs() insertion order, recursive, each dense = (W, b):
  // points; mlp1(l1,l2); ds1; pt1(lin1,q,k,v,pos1,pos2,attn1,attn2,lin2); ds2; pt2;
  // ds3; pt3; mlp2(l1,l2); us1; ptu1; us2; ptu2; us3; ptu3; mlp3(l1,l2)
  const float* points = (const float*)d_in[0];

  Bump ws0{(char*)d_ws, 0, ws_size};

  int ci = 1;
  auto mk = [&](int din, int dout) {
    DenseP p;
    p.W = (const float*)d_in[ci]; p.b = (const float*)d_in[ci + 1]; ci += 2;
    p.din = din; p.dout = dout;
    const int ntiles = cdiv(dout, 16), kcs = cdiv(din, 32);
    const int total = ntiles * kcs * 512;  // 32 lanes * 16 halves per (ntile, kchunk)
    p.Wf = ws0.h((size_t)total);
    prep_wfrag_kernel<<<cdiv(total, 256), 256, 0, stream>>>(p.W, p.Wf, din, dout, total);
    return p;
  };
  auto mkpt = [&](int din, int dim) {
    PTP P;
    P.lin1 = mk(din, dim); P.q = mk(dim, dim); P.k = mk(dim, dim); P.v = mk(dim, dim);
    P.pos1 = mk(3, 8); P.pos2 = mk(8, dim); P.attn1 = mk(dim, 4); P.attn2 = mk(4, dim);
    P.lin2 = mk(dim, dim);
    return P;
  };
  DenseP m1l1 = mk(6, 8), m1l2 = mk(8, 32);
  DenseP ds1 = mk(32, 16);
  PTP pt1 = mkpt(16, 16);
  DenseP ds2 = mk(16, 32);
  PTP pt2 = mkpt(32, 32);
  DenseP ds3 = mk(32, 64);
  PTP pt3 = mkpt(64, 64);
  DenseP m2l1 = mk(64, 8), m2l2 = mk(8, 32);
  DenseP us1 = mk(32, 32);
  PTP ptu1 = mkpt(64, 32);
  DenseP us2 = mk(32, 16);
  PTP ptu2 = mkpt(32, 16);
  DenseP us3 = mk(16, 8);
  PTP ptu3 = mkpt(40, 8);
  DenseP m3l1 = mk(8, 8), m3l2 = mk(8, 13);

  // persistent activation buffers
  float* pos0  = ws0.f((size_t)BATCH * 4096 * 3);
  float* feat0 = ws0.f((size_t)BATCH * 4096 * 6);
  float* out1  = ws0.f((size_t)BATCH * 4096 * 32);
  float* p1 = ws0.f((size_t)BATCH * 1024 * 3);
  float* p2 = ws0.f((size_t)BATCH * 256 * 3);
  float* p3 = ws0.f((size_t)BATCH * 64 * 3);
  float* d1b = ws0.f((size_t)BATCH * 1024 * 16);
  float* d2b = ws0.f((size_t)BATCH * 256 * 32);
  float* d3b = ws0.f((size_t)BATCH * 64 * 64);
  float* uA = ws0.f((size_t)BATCH * 4096 * 40);
  float* uB = ws0.f((size_t)BATCH * 4096 * 40);

  { int total = BATCH * 4096;
    split_points_kernel<<<cdiv(total, 256), 256, 0, stream>>>(points, pos0, feat0, total); }

  { Bump s = ws0; float* h = s.f((size_t)BATCH * 4096 * 8);
    launch_dense(m1l1, feat0, h, BATCH * 4096, 1, stream);
    launch_dense(m1l2, h, out1, BATCH * 4096, 0, stream); }

  { Bump s = ws0; float* f1 = s.f((size_t)BATCH * 1024 * 16);
    run_down(ds1, out1, pos0, 4096, 1024, f1, p1, s, stream);
    run_pt(pt1, f1, p1, 1024, d1b, s, stream); }

  { Bump s = ws0; float* f2 = s.f((size_t)BATCH * 256 * 32);
    run_down(ds2, d1b, p1, 1024, 256, f2, p2, s, stream);
    run_pt(pt2, f2, p2, 256, d2b, s, stream); }

  { Bump s = ws0; float* f3 = s.f((size_t)BATCH * 64 * 64);
    run_down(ds3, d2b, p2, 256, 64, f3, p3, s, stream);
    run_pt(pt3, f3, p3, 64, d3b, s, stream); }

  { Bump s = ws0;
    float* m2 = s.f((size_t)BATCH * 64 * 32);
    { Bump s2 = s; float* h = s2.f((size_t)BATCH * 64 * 8);
      launch_dense(m2l1, d3b, h, BATCH * 64, 1, stream);
      launch_dense(m2l2, h, m2, BATCH * 64, 0, stream); }
    float* u1 = s.f((size_t)BATCH * 256 * 32);
    run_up(us1, m2, p3, p2, 64, 256, u1, s, stream);
    { int total = BATCH * 256 * 64;
      concat_kernel<<<cdiv(total, 256), 256, 0, stream>>>(u1, d2b, uA, 32, 32, total); } }

  { Bump s = ws0; run_pt(ptu1, uA, p2, 256, uB, s, stream); }

  { Bump s = ws0; float* u2 = s.f((size_t)BATCH * 1024 * 16);
    run_up(us2, uB, p2, p1, 256, 1024, u2, s, stream);
    { int total = BATCH * 1024 * 32;
      concat_kernel<<<cdiv(total, 256), 256, 0, stream>>>(u2, d1b, uA, 16, 16, total); } }

  { Bump s = ws0; run_pt(ptu2, uA, p1, 1024, uB, s, stream); }

  { Bump s = ws0; float* u3 = s.f((size_t)BATCH * 4096 * 8);
    run_up(us3, uB, p1, pos0, 1024, 4096, u3, s, stream);
    { int total = BATCH * 4096 * 40;
      concat_kernel<<<cdiv(total, 256), 256, 0, stream>>>(u3, out1, uA, 8, 32, total); } }

  { Bump s = ws0; run_pt(ptu3, uA, pos0, 4096, uB, s, stream); }

  { Bump s = ws0;
    float* h   = s.f((size_t)BATCH * 4096 * 8);
    float* o13 = s.f((size_t)BATCH * 4096 * 13);
    launch_dense(m3l1, uB, h, BATCH * 4096, 1, stream);
    launch_dense(m3l2, h, o13, BATCH * 4096, 0, stream);
    finalnorm_kernel<<<cdiv(BATCH * 4096, 256), 256, 0, stream>>>(o13, (float*)d_out,
                                                                  BATCH * 4096); }
}
